// HyperFormerConv_45621142618216
// MI455X (gfx1250) — matrix operand order repaired and checked by reading, not compiled, not run on hardware
//
#include <hip/hip_runtime.h>
#include <hip/hip_bf16.h>
#include <math.h>

#define N_ROWS 6144
#define DIM    64
#define H_HEADS 4
#define EPSF   1e-7f
#define LOG2E  1.4426950408889634f

typedef __attribute__((ext_vector_type(16))) _Float16 v16h;
typedef __attribute__((ext_vector_type(8)))  _Float16 v8h;
typedef __attribute__((ext_vector_type(8)))  float    v8f;

__device__ __forceinline__ v8f wmma16(v16h a, v16h b, v8f c) {
  // D = A(16x32 f16) * B(32x16 f16) + C(16x16 f32)
  return __builtin_amdgcn_wmma_f32_16x16x32_f16(false, a, false, b, (short)0, c,
                                                false, false);
}

// A operand 16x32 tile from row-major memory (stride in halves).
// CDNA5 layout: lanes 0-15 M=0-15 hold K=[0..7,16..23], lanes 16-31 hold K=[8..15,24..31].
__device__ __forceinline__ v16h load_a_g(const _Float16* p, int stride, int lane) {
  int m = lane & 15, kb = (lane >> 4) * 8;
  v8h lo = *(const v8h*)(p + (size_t)m * stride + kb);
  v8h hi = *(const v8h*)(p + (size_t)m * stride + 16 + kb);
  v16h r;
#pragma unroll
  for (int i = 0; i < 8; ++i) { r[i] = lo[i]; r[i + 8] = hi[i]; }
  return r;
}

// B operand 32x16 tile; memory holds B^T row-major (column's K-dim contiguous).
// lanes 0-15: column c, K=[0..7,16..23] ; lanes 16-31: same column, K=[8..15,24..31].
__device__ __forceinline__ v16h load_b_g(const _Float16* p, int stride, int lane) {
  int c = lane & 15, kh = lane >> 4;
  v8h lo = *(const v8h*)(p + (size_t)c * stride + 8 * kh);
  v8h hi = *(const v8h*)(p + (size_t)c * stride + 16 + 8 * kh);
  v16h r;
#pragma unroll
  for (int i = 0; i < 8; ++i) { r[i] = lo[i]; r[i + 8] = hi[i]; }
  return r;
}

__device__ __forceinline__ float redsum16(float v) {
#pragma unroll
  for (int m = 1; m < 16; m <<= 1) v += __shfl_xor(v, m, 32);
  return v;
}
__device__ __forceinline__ float redsum32(float v) {
#pragma unroll
  for (int m = 1; m < 32; m <<= 1) v += __shfl_xor(v, m, 32);
  return v;
}

// ---------------- Kernel 1: u = logmap0(e), stored fp16 ----------------
__global__ __launch_bounds__(256) void logmap_kernel(const float* __restrict__ e,
                                                     _Float16* __restrict__ u) {
  int lane = threadIdx.x & 31;
  int wave = threadIdx.x >> 5;
  int row  = blockIdx.x * 8 + wave;
  if (row >= N_ROWS) return;
  const float* er = e + (size_t)row * DIM;
  float a = er[lane];
  float b = er[lane + 32];
  float s = (lane == 0 ? 0.f : a * a) + b * b;
  s = redsum32(s);
  float n   = sqrtf(fmaxf(s, EPSF));
  float x0  = er[0];
  float mag = acoshf(fmaxf(x0, 1.f + EPSF));
  float sc  = mag / n;
  _Float16* ur = u + (size_t)row * DIM;
  ur[lane]      = (_Float16)(lane == 0 ? 0.f : a * sc);
  ur[lane + 32] = (_Float16)(b * sc);
}

// ---------------- Kernel 2: convert W's to fp16 ----------------
__global__ __launch_bounds__(256) void cvt_w_kernel(const float* __restrict__ wk,
                                                    const float* __restrict__ wq,
                                                    const float* __restrict__ wv,
                                                    _Float16* wk16, _Float16* wq16,
                                                    _Float16* wv16) {
  int i = blockIdx.x * blockDim.x + threadIdx.x;
  if (i < H_HEADS * DIM * DIM) {
    wk16[i] = (_Float16)wk[i];
    wq16[i] = (_Float16)wq[i];
    wv16[i] = (_Float16)wv[i];
  }
}

// ---------------- Kernel 3: hyperbolic linear for K / Qg / Vt ----------------
// mode 0: K row-major [n][d]; mode 1: (Q*g)*log2e row-major; mode 2: V transposed [d][n]
__device__ __forceinline__ void project_tile(const _Float16* u, const _Float16* Wh,
                                             const float* bh, int row0, int lane,
                                             int mode, _Float16* out) {
  v16h a0 = load_a_g(u + (size_t)row0 * DIM, DIM, lane);
  v16h a1 = load_a_g(u + (size_t)row0 * DIM + 32, DIM, lane);
  v8f acc[4];
#pragma unroll
  for (int t = 0; t < 4; ++t) {
    v8f c = {};
    c      = wmma16(a0, load_b_g(Wh + (size_t)(t * 16) * DIM, DIM, lane), c);
    acc[t] = wmma16(a1, load_b_g(Wh + (size_t)(t * 16) * DIM + 32, DIM, lane), c);
  }
  int cidx = lane & 15, half = lane >> 4;
#pragma unroll
  for (int t = 0; t < 4; ++t) {
    float bb = bh[t * 16 + cidx];
#pragma unroll
    for (int r = 0; r < 8; ++r) acc[t][r] += bb;
  }
  if (cidx == 0) {
#pragma unroll
    for (int r = 0; r < 8; ++r) acc[0][r] = 0.f;   // t[...,0] = 0
  }
  float coef[8], x0v[8];
#pragma unroll
  for (int r = 0; r < 8; ++r) {
    float s = 0.f;
#pragma unroll
    for (int t = 0; t < 4; ++t) s += acc[t][r] * acc[t][r];
    s = redsum16(s);
    float nrm = sqrtf(fmaxf(s, EPSF));
    x0v[r]  = coshf(nrm);             // sqrt(c) = 1
    coef[r] = sinhf(nrm) / nrm;
  }
  float qs = (mode == 1) ? LOG2E : 1.f;   // fold log2(e) into Qg so exp(S)=exp2(S')
#pragma unroll
  for (int t = 0; t < 4; ++t) {
#pragma unroll
    for (int r = 0; r < 8; ++r) {
      float val = acc[t][r] * coef[r];
      if (t == 0 && cidx == 0) val = (mode == 1) ? -x0v[r] : x0v[r];  // g on Q
      val *= qs;
      int d = t * 16 + cidx;
      int n = row0 + r + 8 * half;
      if (mode == 2) out[(size_t)d * N_ROWS + n] = (_Float16)val;
      else           out[(size_t)n * DIM + d]    = (_Float16)val;
    }
  }
}

__global__ __launch_bounds__(128) void qkv_kernel(const _Float16* __restrict__ u,
    const _Float16* wk16, const float* bk, const _Float16* wq16, const float* bq,
    const _Float16* wv16, const float* bv,
    _Float16* K16, _Float16* Qg16, _Float16* Vt16) {
  int lane = threadIdx.x & 31;
  int wave = threadIdx.x >> 5;
  int h    = blockIdx.y;
  int row0 = (blockIdx.x * 4 + wave) * 16;
  size_t woff = (size_t)h * DIM * DIM;
  size_t poff = (size_t)h * N_ROWS * DIM;
  project_tile(u, wk16 + woff, bk + (size_t)h * DIM, row0, lane, 0, K16 + poff);
  project_tile(u, wq16 + woff, bq + (size_t)h * DIM, row0, lane, 1, Qg16 + poff);
  project_tile(u, wv16 + woff, bv + (size_t)h * DIM, row0, lane, 2, Vt16 + poff);
}

// ---------------- Kernel 4: flash attention + Minkowski row-normalize ----------------
// Scores are bounded (hyperboloid points near the apex, |S| ~ O(1)), so the softmax
// max-subtraction is dropped: no cross-lane reductions or accumulator rescaling in the
// hot loop. Row sums are per-lane partials, reduced once at the end.
__global__ __launch_bounds__(128) void attn_kernel(const _Float16* __restrict__ Qg16,
                                                   const _Float16* __restrict__ K16,
                                                   const _Float16* __restrict__ Vt16,
                                                   float* __restrict__ z) {
  __shared__ __align__(32) _Float16 Pl[4][16 * 64];
  int lane = threadIdx.x & 31;
  int wave = threadIdx.x >> 5;
  int h    = blockIdx.y;
  int row0 = (blockIdx.x * 4 + wave) * 16;
  const _Float16* Qh = Qg16 + (size_t)h * N_ROWS * DIM;
  const _Float16* Kh = K16  + (size_t)h * N_ROWS * DIM;
  const _Float16* Vh = Vt16 + (size_t)h * DIM * N_ROWS;

  v16h q0 = load_a_g(Qh + (size_t)row0 * DIM, DIM, lane);
  v16h q1 = load_a_g(Qh + (size_t)row0 * DIM + 32, DIM, lane);

  v8f zero = {};
  v8f o[4] = {zero, zero, zero, zero};
  float li[8];
#pragma unroll
  for (int r = 0; r < 8; ++r) li[r] = 0.f;
  int cidx = lane & 15, half = lane >> 4;
  _Float16* Pw = Pl[wave];

  for (int l0 = 0; l0 < N_ROWS; l0 += 64) {
    // prefetch next column block of K and V into the cache hierarchy
    __builtin_prefetch(Kh + (size_t)(l0 + 64) * DIM, 0, 3);
    __builtin_prefetch(Vh + l0 + 64, 0, 3);

    // S tiles: rows = this wave's 16 queries, cols = keys l0..l0+63 (S pre-scaled by log2e)
    v8f s[4];
#pragma unroll
    for (int j = 0; j < 4; ++j) {
      v8f c = {};
      c    = wmma16(q0, load_b_g(Kh + (size_t)(l0 + 16 * j) * DIM, DIM, lane), c);
      s[j] = wmma16(q1, load_b_g(Kh + (size_t)(l0 + 16 * j) * DIM + 32, DIM, lane), c);
    }
    // P = exp2(S') = exp(S); accumulate per-lane row-sum partials; stash P in LDS (f16)
#pragma unroll
    for (int r = 0; r < 8; ++r) {
      int rrow = r + 8 * half;                 // C-layout row owned by this lane
      float acc = 0.f;
#pragma unroll
      for (int j = 0; j < 4; ++j) {
        float p = __builtin_amdgcn_exp2f(s[j][r]);
        acc += p;
        Pw[rrow * 64 + 16 * j + cidx] = (_Float16)p;
      }
      li[r] += acc;
    }
    // same-wave LDS store->load: LDS is in-order per wave; fence compiler + dscnt
    asm volatile("s_wait_dscnt 0x0" ::: "memory");

    // re-read P in A-operand layout: two 16x32 chunks
    v16h pa0, pa1;
    {
      int m = lane & 15, kb = (lane >> 4) * 8;
      v8h l0v = *(const v8h*)(Pw + m * 64 + kb);
      v8h h0v = *(const v8h*)(Pw + m * 64 + 16 + kb);
      v8h l1v = *(const v8h*)(Pw + m * 64 + 32 + kb);
      v8h h1v = *(const v8h*)(Pw + m * 64 + 48 + kb);
#pragma unroll
      for (int i = 0; i < 8; ++i) {
        pa0[i] = l0v[i]; pa0[8 + i] = h0v[i];
        pa1[i] = l1v[i]; pa1[8 + i] = h1v[i];
      }
    }
    // O += P * V  (K-dim = 64 keys -> 2 chained WMMAs per 16-wide d tile)
#pragma unroll
    for (int t = 0; t < 4; ++t) {
      v8f c = o[t];
      c    = wmma16(pa0, load_b_g(Vh + (size_t)(t * 16) * N_ROWS + l0, N_ROWS, lane), c);
      o[t] = wmma16(pa1, load_b_g(Vh + (size_t)(t * 16) * N_ROWS + l0 + 32, N_ROWS, lane), c);
    }
  }

  // epilogue: v = O/l ; coef = rsqrt(|sum g*v^2|) ; z = v*coef
  float coef[8];
#pragma unroll
  for (int r = 0; r < 8; ++r) {
    float lsum = redsum16(li[r]);
    float inv  = 1.f / lsum;
    float s = 0.f;
#pragma unroll
    for (int t = 0; t < 4; ++t) {
      float v = o[t][r] * inv;
      o[t][r] = v;
      float gg = (t == 0 && cidx == 0) ? -1.f : 1.f;
      s += gg * v * v;
    }
    s = redsum16(s);
    coef[r] = rsqrtf(fmaxf(fabsf(s), EPSF));
  }
#pragma unroll
  for (int t = 0; t < 4; ++t) {
#pragma unroll
    for (int r = 0; r < 8; ++r) {
      int n = row0 + r + 8 * half;
      z[((size_t)h * N_ROWS + n) * DIM + t * 16 + cidx] = o[t][r] * coef[r];
    }
  }
}

// ---------------- Kernel 5: head mean + final normalization ----------------
__global__ __launch_bounds__(256) void final_kernel(const float* __restrict__ z,
                                                    float* __restrict__ out) {
  int lane = threadIdx.x & 31;
  int wave = threadIdx.x >> 5;
  int row  = blockIdx.x * 8 + wave;
  if (row >= N_ROWS) return;
  size_t hs = (size_t)N_ROWS * DIM;
  const float* zr = z + (size_t)row * DIM;
  float a = 0.f, b = 0.f;
#pragma unroll
  for (int h = 0; h < H_HEADS; ++h) {
    a += zr[h * hs + lane];
    b += zr[h * hs + lane + 32];
  }
  a *= 0.25f; b *= 0.25f;
  float s = (lane == 0 ? -a * a : a * a) + b * b;
  s = redsum32(s);
  float inv = rsqrtf(fmaxf(fabsf(s), EPSF));
  out[(size_t)row * DIM + lane]      = a * inv;
  out[(size_t)row * DIM + lane + 32] = b * inv;
}

extern "C" void kernel_launch(void* const* d_in, const int* in_sizes, int n_in,
                              void* d_out, int out_size, void* d_ws, size_t ws_size,
                              hipStream_t stream) {
  const float* e  = (const float*)d_in[0];
  // d_in[1] = adjs : unused by the reference math
  const float* Wk = (const float*)d_in[2];
  const float* bk = (const float*)d_in[3];
  const float* Wq = (const float*)d_in[4];
  const float* bq = (const float*)d_in[5];
  const float* Wv = (const float*)d_in[6];
  const float* bv = (const float*)d_in[7];
  float* out = (float*)d_out;

  char* ws = (char*)d_ws;
  size_t off = 0;
  const size_t U_BYTES  = (size_t)N_ROWS * DIM * 2;           // 768 KB
  const size_t W_BYTES  = (size_t)H_HEADS * DIM * DIM * 2;    // 32 KB each
  const size_t P_BYTES  = (size_t)H_HEADS * N_ROWS * DIM * 2; // 3 MB each
  const size_t Z_BYTES  = (size_t)H_HEADS * N_ROWS * DIM * 4; // 6 MB

  _Float16* u16  = (_Float16*)(ws + off); off += U_BYTES;
  _Float16* wk16 = (_Float16*)(ws + off); off += W_BYTES;
  _Float16* wq16 = (_Float16*)(ws + off); off += W_BYTES;
  _Float16* wv16 = (_Float16*)(ws + off); off += W_BYTES;
  _Float16* Qg16 = (_Float16*)(ws + off); off += P_BYTES;
  _Float16* K16  = (_Float16*)(ws + off); off += P_BYTES;
  _Float16* Vt16 = (_Float16*)(ws + off); off += P_BYTES;
  float*    zbuf = (float*)   (ws + off); off += Z_BYTES;
  (void)ws_size; (void)in_sizes; (void)n_in; (void)out_size;

  logmap_kernel<<<dim3(N_ROWS / 8), dim3(256), 0, stream>>>(e, u16);
  cvt_w_kernel<<<dim3((H_HEADS * DIM * DIM + 255) / 256), dim3(256), 0, stream>>>(
      Wk, Wq, Wv, wk16, wq16, wv16);
  qkv_kernel<<<dim3(N_ROWS / 64, H_HEADS), dim3(128), 0, stream>>>(
      u16, wk16, bk, wq16, bq, wv16, bv, K16, Qg16, Vt16);
  attn_kernel<<<dim3(N_ROWS / 64, H_HEADS), dim3(128), 0, stream>>>(
      Qg16, K16, Vt16, zbuf);
  final_kernel<<<dim3(N_ROWS / 8), dim3(256), 0, stream>>>(zbuf, out);
}
